// TAGNN_Base_13116830122155
// MI455X (gfx1250) — matrix operand order repaired and checked by reading, not compiled, not run on hardware
//
#include <hip/hip_runtime.h>
#include <hip/hip_bf16.h>
#include <math.h>

typedef _Float16 h16;
typedef __attribute__((ext_vector_type(8)))  _Float16 v8h;
typedef __attribute__((ext_vector_type(16))) _Float16 v16h;
typedef __attribute__((ext_vector_type(8)))  float    v8f;

#define BATCH  1024
#define NSEQ   50
#define EMB    100
#define ROWS   (BATCH*NSEQ)     // 51200
#define NITEMS 100000

__device__ __forceinline__ v16h make16(v8h lo, v8h hi) {
  return __builtin_shufflevector(lo, hi, 0,1,2,3,4,5,6,7,8,9,10,11,12,13,14,15);
}

// ---------------- conversion / prep kernels ----------------

// dst[r*blockDim + c] = (r<nrows && c<sk) ? src[r*sk+c] : 0   (f32->f16 pad rows+cols)
// launch: grid = padded row count, block = padded K
__global__ void pad_rows_cols_f16(const float* __restrict__ src, h16* __restrict__ dst,
                                  int nrows, int sk) {
  const int r = blockIdx.x, c = threadIdx.x;
  dst[(size_t)r*blockDim.x + c] =
      (r < nrows && c < sk) ? (h16)src[(size_t)r*sk + c] : (h16)0.f;
}

// gather x = emb_ggnn[seq], and banded adjacency products:
// a_out[i] = x[i] + (i < L-1 ? x[i+1] : 0) ; a_in[i] = x[i] + (1<=i<L ? x[i-1] : 0)
__global__ void gather_adj(const int* __restrict__ seq, const int* __restrict__ lens,
                           const float* __restrict__ embg,
                           h16* __restrict__ x16 /*[ROWS,128]*/,
                           h16* __restrict__ a16 /*[ROWS,224]*/) {
  const int r = blockIdx.x;
  const int b = r / NSEQ, i = r - b*NSEQ;
  const int c = threadIdx.x;               // 256 threads
  const int L = lens[b];
  const int id = seq[r];
  if (c < 128) {
    const float v = (c < EMB) ? embg[(size_t)id*EMB + c] : 0.f;
    x16[(size_t)r*128 + c] = (h16)v;
  }
  if (c < 224) {
    float av = 0.f;
    if (c < EMB) {                          // a_out
      const float xv = embg[(size_t)id*EMB + c];
      const float nx = (i < L-1) ? embg[(size_t)seq[r+1]*EMB + c] : 0.f;
      av = xv + nx;
    } else if (c < 2*EMB) {                 // a_in
      const int cc = c - EMB;
      const float xv = embg[(size_t)id*EMB + cc];
      const float pv = (i >= 1 && i < L) ? embg[(size_t)seq[r-1]*EMB + cc] : 0.f;
      av = xv + pv;
    }
    a16[(size_t)r*224 + c] = (h16)av;
  }
}

// ---- WMMA GEMM, A row-major [M,Kp], B kept N-major [Np,Kp] (== W layout):  C = A*B^T + bias
// Fragments assembled directly from global: per the ISA f16 layouts both A and B
// fragments are contiguous 16B runs, so no LDS transpose / barriers are needed.
// block = 256 threads (8 waves); wave w -> 16 M-rows; grid = (ceil(N/16), M/128).
__global__ void __launch_bounds__(256)
gemm_ant(const h16* __restrict__ A, int lda,
         const h16* __restrict__ Bt,            // [>=ceil16(N) rows, Kp], zero padded
         float* __restrict__ C, int ldc,
         const float* __restrict__ bias,
         int N, int Kp) {
  const int t = threadIdx.x;
  const int w = t >> 5, lane = t & 31;
  const int half = lane >> 4, l16 = lane & 15;
  const int mrow = blockIdx.y*128 + w*16 + l16;
  const int ncol = blockIdx.x*16;

  const h16* arow = A  + (size_t)mrow*lda;
  const h16* brow = Bt + (size_t)(ncol + l16)*Kp;

  v8f acc = {};
  const int nk = Kp >> 5;
  for (int kc = 0; kc < nk; ++kc) {
    v8h alo = *(const v8h*)(arow + kc*32 + half*8);
    v8h ahi = *(const v8h*)(arow + kc*32 + 16 + half*8);
    v8h blo = *(const v8h*)(brow + kc*32 + half*8);
    v8h bhi = *(const v8h*)(brow + kc*32 + 16 + half*8);
    acc = __builtin_amdgcn_wmma_f32_16x16x32_f16(
        false, make16(alo, ahi), false, make16(blo, bhi),
        (short)0, acc, false, false);
  }

  const int col = ncol + l16;
  if (col < N) {
    const float bv = bias ? bias[col] : 0.f;
    const int mb = blockIdx.y*128 + w*16 + half*8;
    #pragma unroll
    for (int r = 0; r < 8; ++r)
      C[(size_t)(mb + r)*ldc + col] = acc[r] + bv;
  }
}

// ---------------- GRU gate fusion ----------------
__global__ void gru_gates(const float* __restrict__ gi, const float* __restrict__ gh,
                          const h16* __restrict__ x16, h16* __restrict__ hg16) {
  const int r = blockIdx.x, c = threadIdx.x;   // 128 threads
  if (c < EMB) {
    const float ir  = gi[(size_t)r*320 + c];
    const float iz  = gi[(size_t)r*320 + 100 + c];
    const float inn = gi[(size_t)r*320 + 200 + c];
    const float hr  = gh[(size_t)r*320 + c];
    const float hz  = gh[(size_t)r*320 + 100 + c];
    const float hn  = gh[(size_t)r*320 + 200 + c];
    const float x   = (float)x16[(size_t)r*128 + c];
    const float rg = 1.f/(1.f + expf(-(ir + hr)));
    const float z  = 1.f/(1.f + expf(-(iz + hz)));
    const float ng = tanhf(inn + rg*hn);
    hg16[(size_t)r*128 + c] = (h16)(ng + z*(x - ng));
  } else {
    hg16[(size_t)r*128 + c] = (h16)0.f;
  }
}

// ---------------- target embedding gather ----------------
__global__ void build_t16(const int* __restrict__ seq, const int* __restrict__ lens,
                          const float* __restrict__ emb_main, h16* __restrict__ t16) {
  const int b = blockIdx.x, c = threadIdx.x;   // 128 threads
  const int tgt = seq[b*NSEQ + lens[b] - 1];
  t16[b*128 + c] = (c < EMB) ? (h16)emb_main[(size_t)tgt*EMB + c] : (h16)0.f;
}

// ---------------- attention pooling + session concat build ----------------
__global__ void attention_pool(const float* __restrict__ h,   // [ROWS,128]
                               const float* __restrict__ q,   // [BATCH,128]
                               const int* __restrict__ seq, const int* __restrict__ lens,
                               const float* __restrict__ emb_main,
                               h16* __restrict__ cat16) {     // [BATCH,320]
  const int b = blockIdx.x, t = threadIdx.x;   // 128 threads
  __shared__ float att[NSEQ];
  __shared__ float qsh[EMB];
  if (t < EMB) qsh[t] = q[b*128 + t];
  __syncthreads();
  if (t < NSEQ) {
    const float* hr = h + (size_t)(b*NSEQ + t)*128;
    float dot = 0.f;
    for (int c = 0; c < EMB; ++c) dot += hr[c]*qsh[c];
    att[t] = dot;
  }
  __syncthreads();
  if (t == 0) {                                // softmax over n=50 (unmasked, per ref)
    float mx = att[0];
    for (int i = 1; i < NSEQ; ++i) mx = fmaxf(mx, att[i]);
    float s = 0.f;
    for (int i = 0; i < NSEQ; ++i) { att[i] = expf(att[i] - mx); s += att[i]; }
    const float inv = 1.f/s;
    for (int i = 0; i < NSEQ; ++i) att[i] *= inv;
  }
  __syncthreads();
  const int L = lens[b];
  const int lastRow = b*NSEQ + L - 1;
  const int tgt = seq[lastRow];
  if (t < EMB) {
    float hg = 0.f;
    for (int i = 0; i < NSEQ; ++i) {
      const float m = (seq[b*NSEQ + i] > 0) ? 1.f : 0.f;   // mask applied post-softmax
      hg += h[(size_t)(b*NSEQ + i)*128 + t] * att[i] * m;
    }
    cat16[b*320 + t]       = (h16)hg;
    cat16[b*320 + 100 + t] = (h16)h[(size_t)lastRow*128 + t];
    cat16[b*320 + 200 + t] = (h16)emb_main[(size_t)tgt*EMB + t];
  } else if (t < 120) {
    cat16[b*320 + 300 + (t - 100)] = (h16)0.f;
  }
}

__global__ void tanh_to_f16(const float* __restrict__ s, h16* __restrict__ s16) {
  const int b = blockIdx.x, c = threadIdx.x;   // 128 threads
  s16[b*128 + c] = (c < EMB) ? (h16)tanhf(s[(size_t)b*128 + c]) : (h16)0.f;
}

// ---------------- final NT GEMM: C[1024,N] = A[1024,128] * Bt[N,128]^T ----------------
// block = 256 threads (8 waves), tile 128(M) x 128(N); A fragments loaded once per wave.
// Output stored non-temporally (409.6 MB streams past L2 so emb16 stays resident);
// next B tile prefetched (global_prefetch_b8).
__global__ void __launch_bounds__(256)
gemm_nt(const h16* __restrict__ A, const h16* __restrict__ Bt,
        float* __restrict__ C, int N) {
  const int t = threadIdx.x;
  const int w = t >> 5, lane = t & 31;
  const int half = lane >> 4, l16 = lane & 15;
  const int mrow = blockIdx.y*128 + w*16 + l16;

  const h16* arow = A + (size_t)mrow*128;
  v16h af[4];
  #pragma unroll
  for (int kc = 0; kc < 4; ++kc) {
    v8h lo = *(const v8h*)(arow + kc*32 + half*8);
    v8h hi = *(const v8h*)(arow + kc*32 + 16 + half*8);
    af[kc] = make16(lo, hi);
  }

  #pragma unroll
  for (int it = 0; it < 8; ++it) {
    const int ncol = blockIdx.x*128 + it*16;
    int nr = ncol + l16; if (nr >= N) nr = N - 1;       // clamp edge reads
    const h16* brow = Bt + (size_t)nr*128;
    if (it < 7) {                                        // prefetch next B tile
      int nr2 = ncol + 16 + l16; if (nr2 >= N) nr2 = N - 1;
      __builtin_prefetch(Bt + (size_t)nr2*128, 0, 1);
    }
    v8f acc = {};
    #pragma unroll
    for (int kc = 0; kc < 4; ++kc) {
      v8h lo = *(const v8h*)(brow + kc*32 + half*8);
      v8h hi = *(const v8h*)(brow + kc*32 + 16 + half*8);
      acc = __builtin_amdgcn_wmma_f32_16x16x32_f16(
          false, af[kc], false, make16(lo, hi), (short)0, acc, false, false);
    }
    const int col = ncol + l16;
    if (col < N) {
      const int mb = blockIdx.y*128 + w*16 + half*8;
      #pragma unroll
      for (int r = 0; r < 8; ++r)
        __builtin_nontemporal_store(acc[r], &C[(size_t)(mb + r)*N + col]);
    }
  }
}

// ---------------- host side ----------------
extern "C" void kernel_launch(void* const* d_in, const int* in_sizes, int n_in,
                              void* d_out, int out_size, void* d_ws, size_t ws_size,
                              hipStream_t stream) {
  (void)in_sizes; (void)n_in; (void)out_size; (void)ws_size;
  const int*   seq      = (const int*)  d_in[0];
  const int*   lens     = (const int*)  d_in[1];
  const float* emb_main = (const float*)d_in[2];
  const float* emb_ggnn = (const float*)d_in[3];
  const float* W_ih     = (const float*)d_in[4];
  const float* W_hh     = (const float*)d_in[5];
  const float* b_ih     = (const float*)d_in[6];
  const float* b_hh     = (const float*)d_in[7];
  const float* W_out_w  = (const float*)d_in[8];
  const float* W_out_b  = (const float*)d_in[9];
  const float* W_w      = (const float*)d_in[10];
  const float* W3_w     = (const float*)d_in[11];
  const float* W3_b     = (const float*)d_in[12];
  float* out = (float*)d_out;

  char* ws = (char*)d_ws;
  size_t off = 0;
  auto carve = [&](size_t bytes) { char* p = ws + off; off += (bytes + 255) & ~(size_t)255; return p; };
  h16*   emb16   = (h16*)  carve((size_t)NITEMS*128*2);  // emb_main f16, K padded to 128
  h16*   x16     = (h16*)  carve((size_t)ROWS*128*2);
  h16*   a16     = (h16*)  carve((size_t)ROWS*224*2);
  h16*   WihP    = (h16*)  carve(320*224*2);             // weights kept [N,K] f16, padded
  h16*   WhhP    = (h16*)  carve(320*128*2);
  h16*   WoutP   = (h16*)  carve(112*128*2);
  h16*   WwP     = (h16*)  carve(112*128*2);
  h16*   W3P     = (h16*)  carve(112*320*2);
  float* gi      = (float*)carve((size_t)ROWS*320*4);
  float* gh      = (float*)carve((size_t)ROWS*320*4);
  h16*   hg16    = (h16*)  carve((size_t)ROWS*128*2);
  float* hfull   = (float*)carve((size_t)ROWS*128*4);
  h16*   t16     = (h16*)  carve((size_t)BATCH*128*2);
  float* qf32    = (float*)carve((size_t)BATCH*128*4);
  h16*   cat16   = (h16*)  carve((size_t)BATCH*320*2);
  float* sessf32 = (float*)carve((size_t)BATCH*128*4);
  h16*   sess16  = (h16*)  carve((size_t)BATCH*128*2);

  // 1) f16 conversions (weights stay [N,K] row-major == native W layout; just pad)
  pad_rows_cols_f16<<<NITEMS, 128, 0, stream>>>(emb_main, emb16, NITEMS, EMB);
  pad_rows_cols_f16<<<320, 224, 0, stream>>>(W_ih,    WihP,  300, 200);
  pad_rows_cols_f16<<<320, 128, 0, stream>>>(W_hh,    WhhP,  300, 100);
  pad_rows_cols_f16<<<112, 128, 0, stream>>>(W_out_w, WoutP, 100, 100);
  pad_rows_cols_f16<<<112, 128, 0, stream>>>(W_w,     WwP,   100, 100);
  pad_rows_cols_f16<<<112, 320, 0, stream>>>(W3_w,    W3P,   100, 300);

  // 2) GGNN inputs
  gather_adj<<<ROWS, 256, 0, stream>>>(seq, lens, emb_ggnn, x16, a16);

  // 3) GRU GEMMs + gates + W_out   (M=51200 -> grid.y=400)
  gemm_ant<<<dim3(19,400), 256, 0, stream>>>(a16,224, WihP, gi,320, b_ih, 300, 224);
  gemm_ant<<<dim3(19,400), 256, 0, stream>>>(x16,128, WhhP, gh,320, b_hh, 300, 128);
  gru_gates<<<ROWS, 128, 0, stream>>>(gi, gh, x16, hg16);
  gemm_ant<<<dim3(7,400), 256, 0, stream>>>(hg16,128, WoutP, hfull,128, W_out_b, 100, 128);

  // 4) attention
  build_t16<<<BATCH, 128, 0, stream>>>(seq, lens, emb_main, t16);
  gemm_ant<<<dim3(7,8), 256, 0, stream>>>(t16,128, WwP, qf32,128, nullptr, 100, 128);
  attention_pool<<<BATCH, 128, 0, stream>>>(hfull, qf32, seq, lens, emb_main, cat16);

  // 5) session embedding
  gemm_ant<<<dim3(7,8), 256, 0, stream>>>(cat16,320, W3P, sessf32,128, W3_b, 100, 320);
  tanh_to_f16<<<BATCH, 128, 0, stream>>>(sessf32, sess16);

  // 6) scores = sess @ emb_main^T   (dominant GEMM; NT output store keeps emb16 in L2)
  gemm_nt<<<dim3((NITEMS+127)/128, 8), 256, 0, stream>>>(sess16, emb16, out, NITEMS);
}
